// Model_11149735100878
// MI455X (gfx1250) — matrix operand order repaired and checked by reading, not compiled, not run on hardware
//
#include <hip/hip_runtime.h>
#include <cstdint>

// ---------------------------------------------------------------------------
// Types for CDNA5 WMMA (wave32). bf16 A/B fragments are 16 x bf16 = 8 VGPRs,
// f32 accumulator is 8 x f32 = 8 VGPRs.
// ---------------------------------------------------------------------------
typedef __attribute__((ext_vector_type(16))) __bf16 v16bf;
typedef __attribute__((ext_vector_type(8)))  float  v8f;
typedef unsigned short u16;
typedef __attribute__((ext_vector_type(8)))  unsigned short u16x8;

__device__ __forceinline__ u16 f2bf(float f) {
    unsigned int x = __float_as_uint(f);
    unsigned int r = x + 0x7FFFu + ((x >> 16) & 1u);   // round-to-nearest-even
    return (u16)(r >> 16);
}

union FragU { v16bf v; u16x8 h[2]; };

__device__ __forceinline__ v16bf load_frag(const u16* p0, const u16* p1) {
    FragU u;
    u.h[0] = *(const u16x8*)p0;   // 16-byte global_load_b128
    u.h[1] = *(const u16x8*)p1;
    return u.v;
}

// ---------------------------------------------------------------------------
// Generic batched GEMM:  C[z] = act( A[z](MxK) * Bt[z](NxK)^T + bias[z] )
// A, Bt are bf16 (u16 bits), K padded to a multiple of 32.
// Register blocking: one wave computes a 16x64 C tile (4 f32 accumulators),
// reusing one A fragment across 4 B fragments per K-step. The 4 independent
// WMMA chains hide the bf16 WMMA->WMMA RAW latency within a single wave,
// which matters for the latency-critical small recurrent GEMMs (M=32).
// 4 waves / 128-thread block span 256 N-columns. Operands stream from global;
// reuse is served by the 192MB L2 (all operands are L2-resident here).
// ---------------------------------------------------------------------------
__global__ void __launch_bounds__(128)
gemm_bt_kernel(const u16* __restrict__ A, const u16* __restrict__ Bt,
               const float* __restrict__ bias, void* __restrict__ Cout,
               int M, int N, int Kp, int lda, int ldb, int ldc,
               long sA, long sB, long sC, long sBias,
               int act, int outBf16)
{
    const int wave = threadIdx.x >> 5;
    const int lane = threadIdx.x & 31;
    const int tn   = blockIdx.x * 4 + wave;
    const int col0 = tn * 64;
    if (col0 >= N) return;                 // uniform per wave: EXEC stays all-1s
    const int row0 = blockIdx.y * 16;
    const long z   = blockIdx.z;

    const u16* a  = A  + z * sA;
    const u16* bt = Bt + z * sB;

    const int g = lane >> 4;               // half-wave group
    const int m = lane & 15;

    int ra = row0 + m; if (ra > M - 1) ra = M - 1;   // clamp: rows beyond M/N are
    // A fragment layout (16x32 bf16): lane g*16+m holds K = g*16 .. g*16+15
    const u16* ap = a + (long)ra * lda + g * 16;

    // B fragment layout (32x16 bf16): lane g*16+n holds K = g*16 .. g*16+15
    const u16* bp[4];
#pragma unroll
    for (int j = 0; j < 4; ++j) {
        int rb = col0 + j * 16 + m;
        if (rb > N - 1) rb = N - 1;        // clamped cols are discarded at store
        bp[j] = bt + (long)rb * ldb + g * 16;
    }

    v8f acc0 = {}, acc1 = {}, acc2 = {}, acc3 = {};
    for (int k0 = 0; k0 < Kp; k0 += 32) {
        __builtin_prefetch(ap + k0 + 64, 0, 1);      // global_prefetch_b8
        v16bf af = load_frag(ap + k0, ap + k0 + 8);
        v16bf b0 = load_frag(bp[0] + k0, bp[0] + k0 + 8);
        v16bf b1 = load_frag(bp[1] + k0, bp[1] + k0 + 8);
        v16bf b2 = load_frag(bp[2] + k0, bp[2] + k0 + 8);
        v16bf b3 = load_frag(bp[3] + k0, bp[3] + k0 + 8);
        acc0 = __builtin_amdgcn_wmma_f32_16x16x32_bf16(false, af, false, b0, (short)0, acc0, false, false);
        acc1 = __builtin_amdgcn_wmma_f32_16x16x32_bf16(false, af, false, b1, (short)0, acc1, false, false);
        acc2 = __builtin_amdgcn_wmma_f32_16x16x32_bf16(false, af, false, b2, (short)0, acc2, false, false);
        acc3 = __builtin_amdgcn_wmma_f32_16x16x32_bf16(false, af, false, b3, (short)0, acc3, false, false);
    }

    // C layout: VGPR r, lanes 0-15 => (M=row0+r, N=col), lanes 16-31 => M=row0+r+8
#pragma unroll
    for (int j = 0; j < 4; ++j) {
        const v8f accv = (j == 0) ? acc0 : (j == 1) ? acc1 : (j == 2) ? acc2 : acc3;
        int col = col0 + j * 16 + m;
        if (col >= N) continue;
        float bc = bias ? bias[z * sBias + col] : 0.f;
#pragma unroll
        for (int r = 0; r < 8; ++r) {
            int row = row0 + r + g * 8;
            if (row < M) {
                float v = accv[r] + bc;
                if (act) v = (v > 0.f) ? v : v * 0.1f;   // leaky_relu(0.1)
                long idx = z * sC + (long)row * ldc + col;
                if (outBf16) ((u16*)Cout)[idx] = f2bf(v);
                else         ((float*)Cout)[idx] = v;
            }
        }
    }
}

// ---------------------------------------------------------------------------
// Elementwise / data-movement kernels
// ---------------------------------------------------------------------------
__global__ void convert_pad_kernel(const float* __restrict__ src, u16* __restrict__ dst,
                                   int rows, int srcK, int dstK)
{
    long i = (long)blockIdx.x * blockDim.x + threadIdx.x;
    long total = (long)rows * dstK;
    if (i >= total) return;
    int r = (int)(i / dstK), k = (int)(i % dstK);
    float v = (k < srcK) ? src[(long)r * srcK + k] : 0.f;
    dst[i] = f2bf(v);
}

__global__ void embed_kernel(const int* __restrict__ words, const int* __restrict__ tags,
                             const float* __restrict__ wemb, const float* __restrict__ temb,
                             u16* __restrict__ x0)
{
    long i = (long)blockIdx.x * blockDim.x + threadIdx.x;
    const int ldx = 224, NE = 100;
    long total = (long)4096 * ldx;
    if (i >= total) return;
    int r = (int)(i / ldx), k = (int)(i % ldx);
    float v = 0.f;
    if (k < NE)          v = wemb[(long)words[r] * NE + k];
    else if (k < 2 * NE) v = temb[(long)tags[r]  * NE + (k - NE)];
    x0[i] = f2bf(v);
}

__global__ void lstm_gate_kernel(const float* __restrict__ gpre, const float* __restrict__ gtmp,
                                 float* __restrict__ c, u16* __restrict__ hbf,
                                 u16* __restrict__ xout, int t)
{
    int i = blockIdx.x * blockDim.x + threadIdx.x;
    const int total = 2 * 32 * 400;
    if (i >= total) return;
    int d = i / (32 * 400);
    int r = i - d * (32 * 400);
    int b = r / 400, j = r - b * 400;
    int s = d ? (127 - t) : t;
    long row = (long)b * 128 + s;
    const float* gp = gpre + (long)d * 4096 * 1600 + row * 1600;
    const float* gt = gtmp + ((long)d * 32 + b) * 1600;
    float gi = gp[j]        + gt[j];
    float gf = gp[400 + j]  + gt[400 + j];
    float gg = gp[800 + j]  + gt[800 + j];
    float go = gp[1200 + j] + gt[1200 + j];
    float si = 1.f / (1.f + __expf(-gi));
    float sf = 1.f / (1.f + __expf(-gf));
    float so = 1.f / (1.f + __expf(-go));
    float tg = tanhf(gg);
    long ci = ((long)d * 32 + b) * 400 + j;
    float cn = sf * c[ci] + si * tg;
    c[ci] = cn;
    float h = so * tanhf(cn);
    u16 hb = f2bf(h);
    hbf[((long)d * 32 + b) * 416 + j] = hb;          // Kp=416-padded h for rec GEMM
    xout[((long)b * 128 + s) * 800 + d * 400 + j] = hb;
}

__global__ void fill_u16_kernel(u16* p, u16 v, long n) {
    long i = (long)blockIdx.x * blockDim.x + threadIdx.x;
    if (i < n) p[i] = v;
}
__global__ void fill_f32_kernel(float* p, float v, long n) {
    long i = (long)blockIdx.x * blockDim.x + threadIdx.x;
    if (i < n) p[i] = v;
}

// xcat[b,s,:400] = x3[b,s,:400]; xcat[b,s,400:] = x3[b,s+1,400:]
__global__ void xcat_kernel(const u16* __restrict__ x3, u16* __restrict__ xcat)
{
    long i = (long)blockIdx.x * blockDim.x + threadIdx.x;
    long total = (long)4064 * 800;
    if (i >= total) return;
    int b = (int)(i / (127 * 800));
    int r = (int)(i - (long)b * 127 * 800);
    int s = r / 800, k = r - s * 800;
    int ss = (k < 400) ? s : (s + 1);
    xcat[i] = x3[((long)b * 128 + ss) * 800 + k];
}

// l1/r1 (4064 x 128 bf16): zeros, col 100 = 1.0; MLP GEMM fills cols 0..99.
__global__ void prefill_ones_kernel(u16* __restrict__ l1, u16* __restrict__ r1)
{
    long i = (long)blockIdx.x * blockDim.x + threadIdx.x;
    long total = (long)4064 * 128;
    if (i >= total) return;
    int k = (int)(i & 127);
    u16 v = (k == 100) ? (u16)0x3F80 : (u16)0;
    l1[i] = v; r1[i] = v;
}

// Wt2[(o*128+j)*128 + i] = W[o,i,j]  (zero-padded j>=101, i>=101)
__global__ void repack_biaffine_kernel(const float* __restrict__ W, u16* __restrict__ Wt2)
{
    long i = (long)blockIdx.x * blockDim.x + threadIdx.x;
    long total = (long)16384 * 128;
    if (i >= total) return;
    int ii = (int)(i & 127);
    long n = i >> 7;
    int o = (int)(n >> 7), j = (int)(n & 127);
    float v = (j < 101 && ii < 101) ? W[((long)o * 101 + ii) * 101 + j] : 0.f;
    Wt2[i] = f2bf(v);
}

// ---------------------------------------------------------------------------
// Host orchestration
// ---------------------------------------------------------------------------
extern "C" void kernel_launch(void* const* d_in, const int* in_sizes, int n_in,
                              void* d_out, int out_size, void* d_ws, size_t ws_size,
                              hipStream_t stream)
{
    (void)in_sizes; (void)n_in; (void)out_size; (void)ws_size;
    const int*   words = (const int*)d_in[0];
    const int*   tags  = (const int*)d_in[1];
    const float* wemb  = (const float*)d_in[2];
    const float* temb  = (const float*)d_in[3];
    const float* Wih0  = (const float*)d_in[4];
    const float* Wih1  = (const float*)d_in[5];
    const float* Wih2  = (const float*)d_in[6];
    const float* Whh   = (const float*)d_in[7];
    const float* bvec  = (const float*)d_in[8];
    const float* mlW   = (const float*)d_in[9];
    const float* mlb   = (const float*)d_in[10];
    const float* mrW   = (const float*)d_in[11];
    const float* mrb   = (const float*)d_in[12];
    const float* biafW = (const float*)d_in[13];
    float* out = (float*)d_out;

    // --- workspace carve-out (256B aligned sub-buffers; all 16B-load safe) ---
    char* base = (char*)d_ws; size_t off = 0;
    auto alloc = [&](size_t bytes) -> char* {
        off = (off + 255) & ~(size_t)255;
        char* p = base + off; off += bytes; return p;
    };
    u16*   x0    = (u16*)  alloc((size_t)4096 * 224 * 2);
    u16*   xo0   = (u16*)  alloc((size_t)4096 * 800 * 2);
    u16*   xo1   = (u16*)  alloc((size_t)4096 * 800 * 2);
    u16*   xo2   = (u16*)  alloc((size_t)4096 * 800 * 2);
    float* gates = (float*)alloc((size_t)2 * 4096 * 1600 * 4);
    float* gtmp  = (float*)alloc((size_t)2 * 32 * 1600 * 4);
    u16*   hbf   = (u16*)  alloc((size_t)2 * 32 * 416 * 2);
    float* cst   = (float*)alloc((size_t)2 * 32 * 400 * 4);
    u16*   Wb0   = (u16*)  alloc((size_t)2 * 1600 * 224 * 2);
    u16*   Wb1   = (u16*)  alloc((size_t)2 * 1600 * 800 * 2);
    u16*   Wb2   = (u16*)  alloc((size_t)2 * 1600 * 800 * 2);
    u16*   Whb   = (u16*)  alloc((size_t)6 * 1600 * 416 * 2);
    u16*   mlb16 = (u16*)  alloc((size_t)100 * 800 * 2);
    u16*   mrb16 = (u16*)  alloc((size_t)100 * 800 * 2);
    u16*   xcat  = (u16*)  alloc((size_t)4064 * 800 * 2);
    u16*   l1    = (u16*)  alloc((size_t)4064 * 128 * 2);
    u16*   r1    = (u16*)  alloc((size_t)4064 * 128 * 2);
    u16*   Wt2   = (u16*)  alloc((size_t)16384 * 128 * 2);
    u16*   Pb    = (u16*)  alloc((size_t)127 * 16384 * 2);

    auto gemm = [&](const u16* A, const u16* Bt, const float* bs, void* C,
                    int M, int N, int Kp, int lda, int ldb, int ldc,
                    long sA, long sB, long sC, long sBias,
                    int act, int obf, int batch) {
        dim3 g((unsigned)((N + 255) / 256), (unsigned)((M + 15) / 16), (unsigned)batch);
        gemm_bt_kernel<<<g, 128, 0, stream>>>(A, Bt, bs, C, M, N, Kp, lda, ldb, ldc,
                                              sA, sB, sC, sBias, act, obf);
    };
    auto cvt = [&](const float* s, u16* dst, int rows, int sk, int dk) {
        long n = (long)rows * dk;
        convert_pad_kernel<<<(unsigned)((n + 255) / 256), 256, 0, stream>>>(s, dst, rows, sk, dk);
    };

    // --- weight conversion to bf16 (K zero-padded to multiples of 32) ---
    cvt(Wih0, Wb0, 3200, 200, 224);
    cvt(Wih1, Wb1, 3200, 800, 800);
    cvt(Wih2, Wb2, 3200, 800, 800);
    cvt(Whh,  Whb, 9600, 400, 416);
    cvt(mlW,  mlb16, 100, 800, 800);
    cvt(mrW,  mrb16, 100, 800, 800);
    {
        long n = (long)16384 * 128;
        repack_biaffine_kernel<<<(unsigned)((n + 255) / 256), 256, 0, stream>>>(biafW, Wt2);
    }

    // --- embedding gather -> bf16 x0 (B*S, 224) ---
    {
        long n = (long)4096 * 224;
        embed_kernel<<<(unsigned)((n + 255) / 256), 256, 0, stream>>>(words, tags, wemb, temb, x0);
    }

    // --- 3 BiLSTM layers ---
    const u16* lin[3]  = { x0, xo0, xo1 };
    u16*       lout[3] = { xo0, xo1, xo2 };
    const u16* lw[3]   = { Wb0, Wb1, Wb2 };
    const int  lK[3]   = { 224, 800, 800 };
    for (int l = 0; l < 3; ++l) {
        // input projection for all timesteps, both directions batched (z=2):
        // gates[d] = x @ Wih[l,d]^T + b[l,d]     (M=4096, N=1600, K=lK)
        gemm(lin[l], lw[l], bvec + (long)l * 3200, gates,
             4096, 1600, lK[l], lK[l], lK[l], 1600,
             0, (long)1600 * lK[l], (long)4096 * 1600, 1600, 0, 0, 2);

        fill_u16_kernel<<<(unsigned)((2 * 32 * 416 + 255) / 256), 256, 0, stream>>>(hbf, 0, 2 * 32 * 416);
        fill_f32_kernel<<<(unsigned)((2 * 32 * 400 + 255) / 256), 256, 0, stream>>>(cst, 0.f, 2 * 32 * 400);

        for (int t = 0; t < 128; ++t) {
            // recurrent GEMM, both directions batched: gtmp[d] = h[d] @ Whh[l,d]^T
            gemm(hbf, Whb + (size_t)l * 2 * 1600 * 416, nullptr, gtmp,
                 32, 1600, 416, 416, 416, 1600,
                 (long)32 * 416, (long)1600 * 416, (long)32 * 1600, 0, 0, 0, 2);
            lstm_gate_kernel<<<100, 256, 0, stream>>>(gates, gtmp, cst, hbf, lout[l], t);
        }
    }

    // --- build xcat = [fwd[:, :-1], bwd[:, 1:]]  (B,127,800) bf16 ---
    {
        long n = (long)4064 * 800;
        xcat_kernel<<<(unsigned)((n + 255) / 256), 256, 0, stream>>>(xo2, xcat);
    }

    // --- MLPs -> l1/r1 with appended ones column, K-padded to 128 ---
    {
        long n = (long)4064 * 128;
        prefill_ones_kernel<<<(unsigned)((n + 255) / 256), 256, 0, stream>>>(l1, r1);
    }
    gemm(xcat, mlb16, mlb, l1, 4064, 100, 800, 800, 800, 128, 0, 0, 0, 0, 1, 1, 1);
    gemm(xcat, mrb16, mrb, r1, 4064, 100, 800, 800, 800, 128, 0, 0, 0, 0, 1, 1, 1);

    // --- biaffine: per batch b ---
    // stage A: P[x, o*128+j] = sum_i l1[b,x,i] * Wt2[(o,j),i]   (M=127,N=16384,K=128)
    // stage B: out[b,x,y,o]  = sum_j r1[b,y,j] * P[x,o,j]       (z=x: M=127,N=128,K=128)
    for (int b = 0; b < 32; ++b) {
        gemm(l1 + (size_t)b * 127 * 128, Wt2, nullptr, Pb,
             127, 16384, 128, 128, 128, 16384, 0, 0, 0, 0, 0, 1, 1);
        gemm(r1 + (size_t)b * 127 * 128, Pb, nullptr, out + (size_t)b * 127 * 127 * 128,
             127, 128, 128, 128, 128, 128,
             0, 16384, (long)127 * 128, 0, 0, 0, 127);
    }
}
